// InvariantPointAttention_56152402428777
// MI455X (gfx1250) — compile-verified, matrix operand order checked by reading, not run on hardware
//
#include <hip/hip_runtime.h>
#include <hip/hip_bf16.h>
#include <math.h>

// ---------------- problem constants ----------------
constexpr int kH  = 12;    // heads
constexpr int kC  = 16;    // head dim
constexpr int kPQ = 4;     // query points
constexpr int kPV = 8;     // value points
constexpr int kCS = 384;   // single-rep channels
constexpr int kCZ = 128;   // pair channels
constexpr int kN  = 768;   // residues

constexpr int kDQ    = kH * kC;        // 192
constexpr int kDQP   = kH * kPQ * 3;   // 144
constexpr int kDVP   = kH * kPV * 3;   // 288
constexpr int kDPROJ = kDQ * 3 + kDQP * 2 + kDVP;   // 1152 = q|k|v|qp|kp|vp
constexpr int kDCAT  = kDQ + kDVP + kH * kCZ;       // 2016

// column offsets inside PRJ rows
constexpr int kOffQ  = 0;
constexpr int kOffK  = kDQ;            // 192
constexpr int kOffV  = kDQ * 2;        // 384
constexpr int kOffQP = kDQ * 3;        // 576
constexpr int kOffKP = kOffQP + kDQP;  // 720
constexpr int kOffVP = kOffKP + kDQP;  // 864

typedef __attribute__((ext_vector_type(16))) __bf16 v16bf;
typedef __attribute__((ext_vector_type(8)))  __bf16 v8bf;
typedef __attribute__((ext_vector_type(8)))  float  v8f;

// 16-bit WMMA A/B operand (CDNA5 ISA 7.12.2): a lane's 16 elements are two
// contiguous K-runs of 8: K = grp*8+0..7 and K = 16+grp*8+0..7. With K the
// fastest-varying dimension this is two 16-byte loads + a shuffle.
__device__ __forceinline__ v16bf load_wmma_row(const __bf16* rowp, int kbase, int grp) {
  v8bf lo = *(const v8bf*)(rowp + kbase + grp * 8);
  v8bf hi = *(const v8bf*)(rowp + kbase + 16 + grp * 8);
  return __builtin_shufflevector(lo, hi, 0, 1, 2, 3, 4, 5, 6, 7,
                                 8, 9, 10, 11, 12, 13, 14, 15);
}

// ---------------- prep kernels: bf16 conversions / weight packing ----------------
__global__ void cvt_s_kernel(const float* __restrict__ in, __bf16* __restrict__ out) {
  int i = blockIdx.x * blockDim.x + threadIdx.x;
  if (i < kN * kCS) out[i] = (__bf16)in[i];
}

// Concatenate the six projection weights into WT[col][k] (K fastest), bf16.
__global__ void build_wcat_kernel(const float* __restrict__ Wq, const float* __restrict__ Wk,
                                  const float* __restrict__ Wv, const float* __restrict__ Wqp,
                                  const float* __restrict__ Wkp, const float* __restrict__ Wvp,
                                  __bf16* __restrict__ WT) {
  int i = blockIdx.x * blockDim.x + threadIdx.x;
  if (i >= kDPROJ * kCS) return;
  int col = i / kCS, k = i % kCS;
  float v;
  if      (col < kOffK)  v = Wq [k * kDQ  + (col - kOffQ)];
  else if (col < kOffV)  v = Wk [k * kDQ  + (col - kOffK)];
  else if (col < kOffQP) v = Wv [k * kDQ  + (col - kOffV)];
  else if (col < kOffKP) v = Wqp[k * kDQP + (col - kOffQP)];
  else if (col < kOffVP) v = Wkp[k * kDQP + (col - kOffKP)];
  else                   v = Wvp[k * kDVP + (col - kOffVP)];
  WT[(size_t)col * kCS + k] = (__bf16)v;
}

// W_out -> WOT[col][k] (K fastest), bf16.
__global__ void cvt_wout_kernel(const float* __restrict__ in, __bf16* __restrict__ out) {
  int i = blockIdx.x * blockDim.x + threadIdx.x;   // i over 2016*384
  if (i < kDCAT * kCS) {
    int k = i / kCS, c = i % kCS;
    out[(size_t)c * kDCAT + k] = (__bf16)in[i];
  }
}

// ---------------- kernel: projection GEMM  PRJ[768,1152] = SB x WT^T -----------
__global__ __launch_bounds__(128)
void proj_gemm_kernel(const __bf16* __restrict__ SB, const __bf16* __restrict__ WT,
                      float* __restrict__ PRJ) {
  const int lane = threadIdx.x & 31;
  const int wave = threadIdx.x >> 5;
  const int tc = blockIdx.x * 4 + wave;            // 72 column tiles, 4 per block
  const int tn = blockIdx.y;                       // 48 row tiles
  const int grp = lane >> 4, lc = lane & 15;

  const __bf16* Arow = SB + (size_t)(tn * 16 + lc) * kCS;
  const __bf16* Brow = WT + (size_t)(tc * 16 + lc) * kCS;
  v8f acc = {};
  #pragma unroll
  for (int kc = 0; kc < kCS / 32; ++kc) {          // 12 K-steps
    v16bf a = load_wmma_row(Arow, kc * 32, grp);
    v16bf b = load_wmma_row(Brow, kc * 32, grp);
    acc = __builtin_amdgcn_wmma_f32_16x16x32_bf16(false, a, false, b,
                                                  (short)0, acc, false, false);
  }
  #pragma unroll
  for (int v = 0; v < 8; ++v)
    PRJ[(size_t)(tn * 16 + v + grp * 8) * kDPROJ + tc * 16 + lc] = acc[v];
}

// ---------------- kernel: point frame transform + norms --------------------------
__global__ __launch_bounds__(256)
void point_fixup_kernel(const float* __restrict__ PRJ, const float* __restrict__ R,
                        const float* __restrict__ T,
                        float* __restrict__ QP, float* __restrict__ KP,
                        float* __restrict__ VP, float* __restrict__ QN,
                        float* __restrict__ KN) {
  const int n = blockIdx.x;
  const int tid = threadIdx.x;
  __shared__ float gqp[kDQP], gkp[kDQP];

  float rr[9], tt[3];
  #pragma unroll
  for (int i = 0; i < 9; ++i) rr[i] = R[(size_t)n * 9 + i];
  #pragma unroll
  for (int i = 0; i < 3; ++i) tt[i] = T[(size_t)n * 3 + i];

  const float* row = PRJ + (size_t)n * kDPROJ;
  // 48 q-points, 48 k-points, 96 v-points : x_glob = R x_loc + t
  for (int pt = tid; pt < 192; pt += 256) {
    if (pt < 48) {
      int base = pt * 3;
      const float* l = row + kOffQP + base;
      #pragma unroll
      for (int i = 0; i < 3; ++i) {
        float g = rr[i*3+0]*l[0] + rr[i*3+1]*l[1] + rr[i*3+2]*l[2] + tt[i];
        gqp[base + i] = g;
        QP[(size_t)n * kDQP + base + i] = g;
      }
    } else if (pt < 96) {
      int base = (pt - 48) * 3;
      const float* l = row + kOffKP + base;
      #pragma unroll
      for (int i = 0; i < 3; ++i) {
        float g = rr[i*3+0]*l[0] + rr[i*3+1]*l[1] + rr[i*3+2]*l[2] + tt[i];
        gkp[base + i] = g;
        KP[(size_t)n * kDQP + base + i] = g;
      }
    } else {
      int base = (pt - 96) * 3;
      const float* l = row + kOffVP + base;
      #pragma unroll
      for (int i = 0; i < 3; ++i) {
        float g = rr[i*3+0]*l[0] + rr[i*3+1]*l[1] + rr[i*3+2]*l[2] + tt[i];
        VP[(size_t)n * kDVP + base + i] = g;
      }
    }
  }
  __syncthreads();
  if (tid < kH) {
    float s1 = 0.f, s2 = 0.f;
    for (int k = 0; k < kPQ * 3; ++k) {
      float a = gqp[tid * 12 + k]; s1 += a * a;
      float b = gkp[tid * 12 + k]; s2 += b * b;
    }
    QN[(size_t)n * kH + tid] = s1;
    KN[(size_t)n * kH + tid] = s2;
  }
}

// ---------------- kernel: fused streaming IPA (single pass over z) ----------------
__global__ __launch_bounds__(512)
void ipa_attn_kernel(const float* __restrict__ Z, const float* __restrict__ R,
                     const float* __restrict__ T, const float* __restrict__ Wb,
                     const float* __restrict__ PRJ,
                     const float* __restrict__ QP, const float* __restrict__ KP,
                     const float* __restrict__ VP, const float* __restrict__ QN,
                     const float* __restrict__ KN, __bf16* __restrict__ CATB) {
  const int n    = blockIdx.x;
  const int tid  = threadIdx.x;
  const int wave = tid >> 5;
  const int lane = tid & 31;
  const int grp  = lane >> 4;
  const int lc   = lane & 15;

  __shared__ __bf16 zt [32][kCZ];       // z tile, zc fastest  (bias A operand)
  __shared__ __bf16 ztT[kCZ][32];       // z tile, m fastest   (o_pair B operand)
  __shared__ __bf16 wbT[16][kCZ];       // W_b^T padded to 16 heads (bias B operand)
  __shared__ float  logitA[kH][32];     // qk/sqrtC - 0.5*Wc*d2
  __shared__ float  biasT[16][32];      // pair bias [h][m]
  __shared__ float  pf [16][32];        // softmax numerators fp32 (rows 12..15 zero)
  __shared__ __bf16 pfb[16][32];        // bf16 copy, m fastest (o_pair A operand)
  __shared__ float  scaleSh[16];
  __shared__ float  runMax[16];
  __shared__ float  runSum[16];
  __shared__ float  qrow[kDQ];
  __shared__ float  qprow[kDQP];
  __shared__ float  qnrow[kH];

  for (int i = tid; i < kDQ;  i += 512) qrow[i]  = PRJ[(size_t)n * kDPROJ + kOffQ + i];
  for (int i = tid; i < kDQP; i += 512) qprow[i] = QP[(size_t)n * kDQP + i];
  // stage W_b transposed + zero-padded (one time, removes divergence from WMMA path)
  for (int i = tid; i < 16 * kCZ; i += 512) {
    int h = i >> 7, k = i & 127;
    wbT[h][k] = (h < kH) ? (__bf16)Wb[k * kH + h] : (__bf16)0.f;
  }
  if (tid < kH) qnrow[tid] = QN[(size_t)n * kH + tid];
  if (tid < 16) { runMax[tid] = -1e30f; runSum[tid] = 0.f; scaleSh[tid] = 1.f; }

  v8f   accP = {};                      // o_pair C tile (waves 0..7: zc tile = wave)
  float acco[6] = {};                   // wave 8: o   (192 = 32 lanes * 6)
  float accp[9] = {};                   // wave 9: o_pts (288 = 32 lanes * 9)
  __syncthreads();

  const float invSqrtC = 0.25f;                    // 1/sqrt(16)
  const float WC = 0.2357022603955158f;            // sqrt(2/(9*PQ))
  const float WL = 0.5773502691896258f;            // sqrt(1/3)

  for (int m0 = 0; m0 < kN; m0 += 32) {
    // ---- stage z tile -> LDS (fp32 -> bf16), packed b128 store + transpose ----
    {
      int i = tid * 8;
      int mrow = i >> 7, col = i & 127;
      const float4* zp = (const float4*)(Z + ((size_t)n * kN + m0 + mrow) * kCZ + col);
      float4 a = zp[0], b = zp[1];
      v8bf pack;
      pack[0] = (__bf16)a.x; pack[1] = (__bf16)a.y;
      pack[2] = (__bf16)a.z; pack[3] = (__bf16)a.w;
      pack[4] = (__bf16)b.x; pack[5] = (__bf16)b.y;
      pack[6] = (__bf16)b.z; pack[7] = (__bf16)b.w;
      *(v8bf*)&zt[mrow][col] = pack;                 // one ds_store_b128
      #pragma unroll
      for (int j = 0; j < 8; ++j) ztT[col + j][mrow] = pack[j];  // transpose scatter
    }
    __syncthreads();

    // ---- phase A: logits (waves 0..11 VALU) + pair bias (waves 12..13 WMMA) ----
    if (wave < 12) {
      int h = wave, m = lane;
      const float* kk = PRJ + (size_t)(m0 + m) * kDPROJ + kOffK + h * kC;
      float qk = 0.f;
      #pragma unroll
      for (int c = 0; c < kC; ++c) qk += qrow[h * kC + c] * kk[c];
      const float* kp = KP + (size_t)(m0 + m) * kDQP + h * 12;
      float cr = 0.f;
      #pragma unroll
      for (int k = 0; k < 12; ++k) cr += qprow[h * 12 + k] * kp[k];
      float sqd = qnrow[h] + KN[(size_t)(m0 + m) * kH + h] - 2.f * cr;
      logitA[h][m] = qk * invSqrtC - 0.5f * WC * sqd;
    } else if (wave < 14) {
      // bias tile: z[m16,k128] x Wb[k128,h16], K split in 4 chunks of 32
      int mofs = (wave - 12) * 16;
      v8f acc = {};
      #pragma unroll
      for (int kc = 0; kc < 4; ++kc) {
        v16bf a = load_wmma_row(&zt[mofs + lc][0], kc * 32, grp);
        v16bf b = load_wmma_row(&wbT[lc][0],       kc * 32, grp);
        acc = __builtin_amdgcn_wmma_f32_16x16x32_bf16(false, a, false, b,
                                                      (short)0, acc, false, false);
      }
      #pragma unroll
      for (int v = 0; v < 8; ++v) biasT[lc][mofs + v + grp * 8] = acc[v];
    } else if (wave == 14) {
      // prefetch next z tile (one cacheline start per lane)
      int m1 = m0 + 32;
      if (m1 < kN)
        __builtin_prefetch(Z + ((size_t)n * kN + m1 + lane) * kCZ, 0, 1);
    }
    __syncthreads();

    // ---- phase B: online softmax update (wave 15) ----
    if (wave == 15) {
      if (lane < kH) {
        int h = lane;
        float mx = runMax[h];
        for (int m = 0; m < 32; ++m) {
          float lg = WL * (logitA[h][m] + biasT[h][m]);
          mx = fmaxf(mx, lg);
        }
        float scale = __expf(runMax[h] - mx);
        float s = 0.f;
        for (int m = 0; m < 32; ++m) {
          float lg = WL * (logitA[h][m] + biasT[h][m]);
          float p  = __expf(lg - mx);
          pf[h][m]  = p;
          pfb[h][m] = (__bf16)p;
          s += p;
        }
        runSum[h] = runSum[h] * scale + s;
        runMax[h] = mx;
        scaleSh[h] = scale;
      } else if (lane < 16) {
        for (int m = 0; m < 32; ++m) { pf[lane][m] = 0.f; pfb[lane][m] = (__bf16)0.f; }
        scaleSh[lane] = 1.f;
      }
    }
    __syncthreads();

    // ---- phase C: accumulate (o_pair WMMA, o / o_pts VALU) ----
    if (wave < 8) {
      #pragma unroll
      for (int v = 0; v < 8; ++v) accP[v] *= scaleSh[v + grp * 8];
      v16bf a = load_wmma_row(&pfb[lc][0],             0, grp);  // A: p[h16, m32]
      v16bf b = load_wmma_row(&ztT[wave * 16 + lc][0], 0, grp);  // B: z[m32, zc16]
      accP = __builtin_amdgcn_wmma_f32_16x16x32_bf16(false, a, false, b,
                                                     (short)0, accP, false, false);
    } else if (wave == 8) {
      #pragma unroll
      for (int j = 0; j < 6; ++j) {
        int idx = lane * 6 + j;
        int h = idx >> 4;
        float s = acco[j] * scaleSh[h];
        for (int m = 0; m < 32; ++m)
          s += pf[h][m] * PRJ[(size_t)(m0 + m) * kDPROJ + kOffV + idx];
        acco[j] = s;
      }
    } else if (wave == 9) {
      #pragma unroll
      for (int j = 0; j < 9; ++j) {
        int idx = lane * 9 + j;
        int h = idx / 24;
        float s = accp[j] * scaleSh[h];
        for (int m = 0; m < 32; ++m)
          s += pf[h][m] * VP[(size_t)(m0 + m) * kDVP + idx];
        accp[j] = s;
      }
    }
    __syncthreads();
  }

  // ---- epilogue: normalize, inverse frame transform, emit bf16 cat row ----
  __bf16* cat = CATB + (size_t)n * kDCAT;
  if (wave < 8) {
    #pragma unroll
    for (int v = 0; v < 8; ++v) {
      int h = v + grp * 8;
      if (h < kH)
        cat[kDQ + kDVP + h * kCZ + wave * 16 + lc] = (__bf16)(accP[v] / runSum[h]);
    }
  } else if (wave == 8) {
    #pragma unroll
    for (int j = 0; j < 6; ++j) {
      int idx = lane * 6 + j;
      int h = idx >> 4;
      cat[idx] = (__bf16)(acco[j] / runSum[h]);
    }
  } else if (wave == 9) {
    float rr[9], tt[3];
    #pragma unroll
    for (int i = 0; i < 9; ++i) rr[i] = R[(size_t)n * 9 + i];
    #pragma unroll
    for (int i = 0; i < 3; ++i) tt[i] = T[(size_t)n * 3 + i];
    #pragma unroll
    for (int pj = 0; pj < 3; ++pj) {                 // 3 points per lane
      int base = lane * 9 + pj * 3;
      int h = base / 24;
      float g[3];
      #pragma unroll
      for (int i = 0; i < 3; ++i) g[i] = accp[pj * 3 + i] / runSum[h] - tt[i];
      #pragma unroll
      for (int jc = 0; jc < 3; ++jc) {               // R^T (x - t)
        float o = rr[0 * 3 + jc] * g[0] + rr[1 * 3 + jc] * g[1] + rr[2 * 3 + jc] * g[2];
        cat[kDQ + base + jc] = (__bf16)o;
      }
    }
  }
}

// ---------------- kernel: output projection [768,2016] x [2016,384] ----------------
__global__ __launch_bounds__(128)
void out_gemm_kernel(const __bf16* __restrict__ A, const __bf16* __restrict__ BwT,
                     const float* __restrict__ bias, float* __restrict__ OUT) {
  const int lane = threadIdx.x & 31;
  const int wave = threadIdx.x >> 5;
  const int tc = blockIdx.x * 4 + wave;            // 24 column tiles, 4 per block
  const int tn = blockIdx.y;                       // 48 row tiles
  const int grp = lane >> 4, lc = lane & 15;

  const __bf16* Arow = A   + (size_t)(tn * 16 + lc) * kDCAT;  // K fastest
  const __bf16* Brow = BwT + (size_t)(tc * 16 + lc) * kDCAT;  // K fastest (transposed W)
  v8f acc = {};
  for (int kc = 0; kc < kDCAT / 32; ++kc) {        // 63 K-steps
    v16bf a = load_wmma_row(Arow, kc * 32, grp);
    v16bf b = load_wmma_row(Brow, kc * 32, grp);
    acc = __builtin_amdgcn_wmma_f32_16x16x32_bf16(false, a, false, b,
                                                  (short)0, acc, false, false);
  }
  #pragma unroll
  for (int v = 0; v < 8; ++v) {
    int nrow = tn * 16 + v + grp * 8;
    int col  = tc * 16 + lc;
    OUT[(size_t)nrow * kCS + col] = acc[v] + bias[col];
  }
}

// ---------------- launch ----------------
extern "C" void kernel_launch(void* const* d_in, const int* in_sizes, int n_in,
                              void* d_out, int out_size, void* d_ws, size_t ws_size,
                              hipStream_t stream) {
  const float* S    = (const float*)d_in[0];
  const float* Z    = (const float*)d_in[1];
  const float* R    = (const float*)d_in[2];
  const float* T    = (const float*)d_in[3];
  const float* Wq   = (const float*)d_in[4];
  const float* Wk   = (const float*)d_in[5];
  const float* Wv   = (const float*)d_in[6];
  const float* Wqp  = (const float*)d_in[7];
  const float* Wkp  = (const float*)d_in[8];
  const float* Wvp  = (const float*)d_in[9];
  const float* Wb   = (const float*)d_in[10];
  const float* Wout = (const float*)d_in[11];
  const float* bout = (const float*)d_in[12];
  float* OUT = (float*)d_out;

  char* ws = (char*)d_ws;
  size_t off = 0;
  auto alloc = [&](size_t bytes) -> char* {
    char* p = ws + off;
    off += (bytes + 255) & ~(size_t)255;
    return p;
  };
  float*  PRJ  = (float*)alloc((size_t)kN * kDPROJ * 4);
  float*  QP   = (float*)alloc((size_t)kN * kDQP * 4);
  float*  KP   = (float*)alloc((size_t)kN * kDQP * 4);
  float*  VP   = (float*)alloc((size_t)kN * kDVP * 4);
  float*  QN   = (float*)alloc((size_t)kN * kH * 4);
  float*  KN   = (float*)alloc((size_t)kN * kH * 4);
  __bf16* SB   = (__bf16*)alloc((size_t)kN * kCS * 2);
  __bf16* WCT  = (__bf16*)alloc((size_t)kDPROJ * kCS * 2);
  __bf16* CATB = (__bf16*)alloc((size_t)kN * kDCAT * 2);
  __bf16* WOT  = (__bf16*)alloc((size_t)kDCAT * kCS * 2);

  cvt_s_kernel<<<(kN * kCS + 255) / 256, 256, 0, stream>>>(S, SB);
  build_wcat_kernel<<<(kDPROJ * kCS + 255) / 256, 256, 0, stream>>>(
      Wq, Wk, Wv, Wqp, Wkp, Wvp, WCT);
  cvt_wout_kernel<<<(kDCAT * kCS + 255) / 256, 256, 0, stream>>>(Wout, WOT);
  proj_gemm_kernel<<<dim3(kDPROJ / 16 / 4, kN / 16), 128, 0, stream>>>(SB, WCT, PRJ);
  point_fixup_kernel<<<kN, 256, 0, stream>>>(PRJ, R, T, QP, KP, VP, QN, KN);
  ipa_attn_kernel<<<kN, 512, 0, stream>>>(Z, R, T, Wb, PRJ, QP, KP, VP, QN, KN, CATB);
  out_gemm_kernel<<<dim3(kCS / 16 / 4, kN / 16), 128, 0, stream>>>(CATB, WOT, bout, OUT);
}
